// RelLearnableDecoderLayer_51917564674337
// MI455X (gfx1250) — matrix operand order repaired and checked by reading, not compiled, not run on hardware
//
#include <hip/hip_runtime.h>
#include <cstdint>
#include <cstddef>

// ---------------------------------------------------------------------------
// Transformer-XL RelLearnableDecoderLayer for MI455X (gfx1250, wave32, WMMA).
// Compute-bound (~90 GFLOP vs ~80MB traffic @23.3TB/s) => f16 WMMA f32-accum.
// ---------------------------------------------------------------------------

typedef _Float16 f16;
typedef _Float16 v16h __attribute__((ext_vector_type(16)));
typedef float    v8f  __attribute__((ext_vector_type(8)));

#define QLEN   1024
#define MLEN   1024
#define KLEN   2048
#define DMODEL 1024
#define NH     16
#define DHD    64
#define NB     2
#define DINNER 4096
#define ATT_SCALE 0.125f

// ---- WMMA fragment helpers (contiguous k order, same for A and B) ------------------
// A-matrix 16x32 f16: lane(row=lane&15): halves 0-15 = K[h16 .. h16+15] (contiguous 32B)
__device__ __forceinline__ v16h frag_ilv(const f16* p) {
  v16h f;
  __builtin_memcpy(&f, p, 32);
  return f;
}
// B-matrix 32x16 f16: lane(col=lane&15): halves 0-15 = K[h16 .. h16+15] (contiguous 32B)
__device__ __forceinline__ v16h frag_ctg(const f16* p) {
  v16h f;
  __builtin_memcpy(&f, p, 32);
  return f;
}
__device__ __forceinline__ v8f wmma32(v16h a, v16h b, v8f c) {
  return __builtin_amdgcn_wmma_f32_16x16x32_f16(false, a, false, b, (short)0, c,
                                                false, false);
}
__device__ __forceinline__ v8f v8zero() {
  v8f z;
  #pragma unroll
  for (int e = 0; e < 8; ++e) z[e] = 0.f;
  return z;
}

// ---------------------------------------------------------------------------
// Elementwise conversion / packing kernels
// ---------------------------------------------------------------------------
__global__ void cvt16_kernel(const float* __restrict__ src, f16* __restrict__ dst,
                             size_t n) {
  for (size_t i = (size_t)blockIdx.x * blockDim.x + threadIdx.x; i < n;
       i += (size_t)gridDim.x * blockDim.x)
    dst[i] = (f16)src[i];
}

// cat = [mems; dec_inp] flattened rows (kk*B + b), f32 -> f16
__global__ void build_cat_kernel(const float* __restrict__ mems,
                                 const float* __restrict__ dec_inp,
                                 f16* __restrict__ cat16) {
  size_t n = (size_t)KLEN * NB * DMODEL;
  for (size_t i = (size_t)blockIdx.x * blockDim.x + threadIdx.x; i < n;
       i += (size_t)gridDim.x * blockDim.x) {
    size_t row = i >> 10;            // /DMODEL
    int    c   = (int)(i & 1023);
    int    kk  = (int)(row >> 1);
    int    b   = (int)(row & 1);
    float  v;
    if (kk < MLEN) v = mems[((size_t)kk * NB + b) * DMODEL + c];
    else           v = dec_inp[((size_t)(kk - MLEN) * NB + b) * DMODEL + c];
    cat16[i] = (f16)v;
  }
}

// r_emb [K,H,DH] f32 -> re16 [H][K][DH] f16 (head-major rows for B-fragments)
__global__ void remb_kernel(const float* __restrict__ r_emb, f16* __restrict__ re16) {
  size_t n = (size_t)KLEN * NH * DHD;
  for (size_t i = (size_t)blockIdx.x * blockDim.x + threadIdx.x; i < n;
       i += (size_t)gridDim.x * blockDim.x) {
    int s = (int)(i / (NH * DHD));
    int rest = (int)(i % (NH * DHD));
    int h = rest >> 6, d = rest & 63;
    re16[((size_t)h * KLEN + s) * DHD + d] = (f16)r_emb[i];
  }
}

// heads16 [K*B, 3*H*DH] -> q16/q16b [B][H][Q][DH], k16 [B][H][K][DH], vT16 [B][H][DH][K]
__global__ void split_heads_kernel(const f16* __restrict__ heads,
                                   const float* __restrict__ r_w_bias,
                                   f16* __restrict__ q16, f16* __restrict__ q16b,
                                   f16* __restrict__ k16, f16* __restrict__ vT16) {
  size_t n = (size_t)KLEN * NB * 3 * NH * DHD;
  for (size_t i = (size_t)blockIdx.x * blockDim.x + threadIdx.x; i < n;
       i += (size_t)gridDim.x * blockDim.x) {
    size_t row = i / 3072;
    int c = (int)(i % 3072);
    int kk = (int)(row >> 1), b = (int)(row & 1);
    int sec = c >> 10, cc = c & 1023;
    int h = cc >> 6, d = cc & 63;
    f16 hv = heads[i];
    if (sec == 0) {
      if (kk >= MLEN) {
        int qi = kk - MLEN;
        size_t o = (((size_t)b * NH + h) * QLEN + qi) * DHD + d;
        q16[o]  = hv;
        q16b[o] = (f16)((float)hv + r_w_bias[h * DHD + d]);
      }
    } else if (sec == 1) {
      k16[(((size_t)b * NH + h) * KLEN + kk) * DHD + d] = hv;
    } else {
      vT16[(((size_t)b * NH + h) * DHD + d) * KLEN + kk] = hv;
    }
  }
}

// ---------------------------------------------------------------------------
// Generic NT WMMA GEMM: C[M,N](f32/f16) = A16[M,Kd] * W16[N,Kd]^T (+bias,+relu)
// 128x128 block, BK=32, 256 threads = 8 waves, wave tile 32x64.
// ---------------------------------------------------------------------------
#define GLDA 40   // 32 + 8 halves pad -> 80B row stride (16B aligned, no bank conflicts)

__global__ __launch_bounds__(256) void gemm_nt_kernel(
    const f16* __restrict__ A, const f16* __restrict__ W,
    const float* __restrict__ bias, float* __restrict__ Cf, f16* __restrict__ Ch,
    int Mr, int N, int Kd, int relu) {
  __shared__ f16 sA[128 * GLDA];
  __shared__ f16 sB[128 * GLDA];
  const int tid = threadIdx.x;
  const int wv = tid >> 5, lane = tid & 31, hl = lane >> 4, r = lane & 15;
  const int wm = wv >> 1, wn = wv & 1;
  const int rowBase = blockIdx.y * 128;
  const int colBase = blockIdx.x * 128;

  v8f acc[2][4];
  #pragma unroll
  for (int i = 0; i < 2; ++i)
    #pragma unroll
    for (int j = 0; j < 4; ++j) acc[i][j] = v8zero();

  for (int kb = 0; kb < Kd; kb += 32) {
    #pragma unroll
    for (int p = 0; p < 2; ++p) {
      int id = p * 256 + tid;
      int row = id >> 2, ch = id & 3;
      const f16* ga = A + (size_t)(rowBase + row) * Kd + kb + ch * 8;
      const f16* gb = W + (size_t)(colBase + row) * Kd + kb + ch * 8;
      if (kb + 32 < Kd) { __builtin_prefetch(ga + 32, 0, 0); __builtin_prefetch(gb + 32, 0, 0); }
      __builtin_memcpy(&sA[row * GLDA + ch * 8], ga, 16);
      __builtin_memcpy(&sB[row * GLDA + ch * 8], gb, 16);
    }
    __syncthreads();
    v16h af[2], bf[4];
    #pragma unroll
    for (int fm = 0; fm < 2; ++fm)
      af[fm] = frag_ilv(&sA[(wm * 32 + fm * 16 + r) * GLDA + hl * 16]);
    #pragma unroll
    for (int fn = 0; fn < 4; ++fn)
      bf[fn] = frag_ctg(&sB[(wn * 64 + fn * 16 + r) * GLDA + hl * 16]);
    #pragma unroll
    for (int fm = 0; fm < 2; ++fm)
      #pragma unroll
      for (int fn = 0; fn < 4; ++fn)
        acc[fm][fn] = wmma32(af[fm], bf[fn], acc[fm][fn]);
    __syncthreads();
  }

  // epilogue: C/D layout -> m = g + 8*half, n = lane&15
  #pragma unroll
  for (int fm = 0; fm < 2; ++fm) {
    int grow0 = rowBase + wm * 32 + fm * 16 + hl * 8;
    #pragma unroll
    for (int fn = 0; fn < 4; ++fn) {
      int col = colBase + wn * 64 + fn * 16 + r;
      float bv = bias ? bias[col] : 0.f;
      #pragma unroll
      for (int g = 0; g < 8; ++g) {
        float v = acc[fm][fn][g] + bv;
        if (relu) v = v > 0.f ? v : 0.f;
        size_t o = (size_t)(grow0 + g) * N + col;
        if (Cf) Cf[o] = v;
        if (Ch) Ch[o] = (f16)v;
      }
    }
  }
}

// ---------------------------------------------------------------------------
// Fused relative attention: per (b, h, 64-query block), 4 waves x 16 queries.
// Score buffer 64x2048 f16 lives in LDS (257KB of the WGP's 320KB).
// BD via shifted B_ s-tiles with ping-pong reuse (window slides by 16).
// ---------------------------------------------------------------------------
#define SCS 2056   // 2048 + 8 halves pad (row stride 4112B, 16B aligned)
#define ATT_SMEM (64 * SCS * 2 + 4 * 2 * 16 * 18 * 4 + 64 * 4)

__global__ __launch_bounds__(128) void attn_kernel(
    const f16* __restrict__ q16, const f16* __restrict__ q16b,
    const f16* __restrict__ k16, const f16* __restrict__ vT16,
    const f16* __restrict__ re16, const float* __restrict__ r_bias,
    f16* __restrict__ vec16) {
  extern __shared__ char smem[];
  f16*   sc     = (f16*)smem;
  float* strip  = (float*)(smem + 64 * SCS * 2);
  float* rowsum = strip + 4 * 2 * 16 * 18;

  const int wv = threadIdx.x >> 5, lane = threadIdx.x & 31;
  const int hl = lane >> 4, r = lane & 15;
  const int b = blockIdx.z, h = blockIdx.y;
  const int iq = blockIdx.x * 64 + wv * 16;     // wave's first query (global)
  float* wstrip = strip + wv * (2 * 16 * 18);

  // Q fragments (A-operand), both biased (AC) and plain (B_)
  const f16* qb_base = q16b + (((size_t)b * NH + h) * QLEN + iq) * DHD;
  const f16* qp_base = q16  + (((size_t)b * NH + h) * QLEN + iq) * DHD;
  v16h qa[2], qn[2];
  #pragma unroll
  for (int c = 0; c < 2; ++c) {
    qa[c] = frag_ilv(qb_base + r * DHD + c * 32 + hl * 16);
    qn[c] = frag_ilv(qp_base + r * DHD + c * 32 + hl * 16);
  }

  // B_ s-tile: [iq..iq+15] x [st..st+15] over shifted position axis s
  auto btile = [&](int st) -> v8f {
    v8f bt = v8zero();
    int srow = st + r; if (srow > KLEN - 1) srow = KLEN - 1;   // clamp (masked anyway)
    #pragma unroll
    for (int c = 0; c < 2; ++c) {
      v16h rb = frag_ctg(re16 + ((size_t)h * KLEN + srow) * DHD + c * 32 + hl * 16);
      bt = wmma32(qn[c], rb, bt);
    }
    return bt;
  };
  auto store_strip = [&](v8f bt, int par) {
    float* dst = wstrip + par * 288;
    #pragma unroll
    for (int g = 0; g < 8; ++g) dst[(g + hl * 8) * 18 + r] = bt[g];
  };

  // prologue: low tile of the first shift window  (s_low(j0=0) = 1008 - iq)
  int st0 = QLEN - 16 - iq;                 // multiple of 16, >= 0
  store_strip(btile(st0), (st0 >> 4) & 1);

  // ---- score phase -------------------------------------------------------
  for (int j0 = 0; j0 < KLEN; j0 += 16) {
    if (j0 > iq + 15 + MLEN) {              // fully-masked tile
      #pragma unroll
      for (int g = 0; g < 8; ++g)
        sc[(size_t)(wv * 16 + g + hl * 8) * SCS + j0 + r] = (f16)(-__builtin_inff());
      continue;
    }
    v8f ac = v8zero();
    #pragma unroll
    for (int c = 0; c < 2; ++c) {
      v16h kf = frag_ctg(k16 + (((size_t)b * NH + h) * KLEN + j0 + r) * DHD +
                         c * 32 + hl * 16);
      ac = wmma32(qa[c], kf, ac);
    }
    int stn  = j0 + QLEN - iq;              // new high s-tile of the window
    int parn = (stn >> 4) & 1;
    store_strip(btile(stn), parn);
    asm volatile("s_wait_dscnt 0" ::: "memory");
    const float* oldb = wstrip + (parn ^ 1) * 288;
    const float* newb = wstrip + parn * 288;
    #pragma unroll
    for (int g = 0; g < 8; ++g) {
      int m  = g + hl * 8;
      int ig = iq + m;
      int jg = j0 + r;
      int rel = 15 + r - m;                 // 0..30 within window
      float bv = (rel < 16) ? oldb[m * 18 + rel] : newb[m * 18 + rel - 16];
      int s = jg + QLEN - 1 - ig; if (s > KLEN - 1) s = KLEN - 1;
      float scv = (ac[g] + bv + r_bias[(size_t)s * NH + h]) * ATT_SCALE;
      if (jg > ig + MLEN) scv = -__builtin_inff();
      sc[(size_t)(wv * 16 + m) * SCS + jg] = (f16)scv;
    }
  }
  asm volatile("s_wait_dscnt 0" ::: "memory");

  // ---- softmax (two-pass over LDS, wave32 shuffles) ----------------------
  #pragma unroll 1
  for (int rr = 0; rr < 16; ++rr) {
    int row = wv * 16 + rr;
    f16* wrow = sc + (size_t)row * SCS;
    float mx = -__builtin_inff();
    for (int t = lane; t < KLEN; t += 32) {
      float xv = (float)wrow[t]; mx = xv > mx ? xv : mx;
    }
    #pragma unroll
    for (int o2 = 16; o2; o2 >>= 1) {
      float ot = __shfl_xor(mx, o2, 32); mx = ot > mx ? ot : mx;
    }
    float sum = 0.f;
    for (int t = lane; t < KLEN; t += 32) {
      float e = __expf((float)wrow[t] - mx);
      wrow[t] = (f16)e;
      sum += e;
    }
    #pragma unroll
    for (int o2 = 16; o2; o2 >>= 1) sum += __shfl_xor(sum, o2, 32);
    if (lane == 0) rowsum[row] = sum;
  }
  asm volatile("s_wait_dscnt 0" ::: "memory");

  // ---- P·V phase ---------------------------------------------------------
  v8f o[4];
  #pragma unroll
  for (int dt = 0; dt < 4; ++dt) o[dt] = v8zero();
  int jmax = iq + 16 + MLEN; if (jmax > KLEN) jmax = KLEN;
  jmax = (jmax + 31) & ~31;
  for (int jc = 0; jc < jmax; jc += 32) {
    v16h pf = frag_ilv(sc + (size_t)(wv * 16 + r) * SCS + jc + hl * 16);
    #pragma unroll
    for (int dt = 0; dt < 4; ++dt) {
      v16h vf = frag_ctg(vT16 + (((size_t)b * NH + h) * DHD + dt * 16 + r) * KLEN +
                         jc + hl * 16);
      o[dt] = wmma32(pf, vf, o[dt]);
    }
  }
  // normalize + write vec16 [Q*B, H*DH]
  #pragma unroll
  for (int dt = 0; dt < 4; ++dt)
    #pragma unroll
    for (int g = 0; g < 8; ++g) {
      int m  = g + hl * 8;
      int ig = iq + m;
      float inv = 1.f / rowsum[wv * 16 + m];
      vec16[((size_t)ig * NB + b) * DMODEL + h * DHD + dt * 16 + r] =
          (f16)(o[dt][g] * inv);
    }
}

// ---------------------------------------------------------------------------
// Residual + LayerNorm over D=1024 (one 256-thread block per row)
// ---------------------------------------------------------------------------
__global__ __launch_bounds__(256) void add_ln_kernel(
    const float* __restrict__ x, const float* __restrict__ y,
    const float* __restrict__ g, const float* __restrict__ beta,
    float* __restrict__ outf, f16* __restrict__ out16) {
  __shared__ float red[256];
  const int row = blockIdx.x, tid = threadIdx.x;
  float v[4]; float s = 0.f;
  #pragma unroll
  for (int q = 0; q < 4; ++q) {
    int c = tid + q * 256;
    v[q] = x[(size_t)row * DMODEL + c] + y[(size_t)row * DMODEL + c];
    s += v[q];
  }
  red[tid] = s; __syncthreads();
  for (int off = 128; off; off >>= 1) {
    if (tid < off) red[tid] += red[tid + off];
    __syncthreads();
  }
  float mean = red[0] * (1.f / DMODEL); __syncthreads();
  float s2 = 0.f;
  #pragma unroll
  for (int q = 0; q < 4; ++q) { float d = v[q] - mean; s2 += d * d; }
  red[tid] = s2; __syncthreads();
  for (int off = 128; off; off >>= 1) {
    if (tid < off) red[tid] += red[tid + off];
    __syncthreads();
  }
  float inv = rsqrtf(red[0] * (1.f / DMODEL) + 1e-5f);
  #pragma unroll
  for (int q = 0; q < 4; ++q) {
    int c = tid + q * 256;
    float o = (v[q] - mean) * inv * g[c] + beta[c];
    if (outf)  outf[(size_t)row * DMODEL + c] = o;
    if (out16) out16[(size_t)row * DMODEL + c] = (f16)o;
  }
}

// ---------------------------------------------------------------------------
// Host-side orchestration
// ---------------------------------------------------------------------------
extern "C" void kernel_launch(void* const* d_in, const int* in_sizes, int n_in,
                              void* d_out, int out_size, void* d_ws, size_t ws_size,
                              hipStream_t stream) {
  (void)in_sizes; (void)n_in; (void)out_size; (void)ws_size;
  const float* dec_inp  = (const float*)d_in[0];
  const float* mems     = (const float*)d_in[1];
  const float* r_emb    = (const float*)d_in[2];
  const float* r_w_bias = (const float*)d_in[3];
  const float* r_bias   = (const float*)d_in[4];
  const float* qkv_w    = (const float*)d_in[5];
  const float* o_w      = (const float*)d_in[6];
  const float* ln1_g    = (const float*)d_in[7];
  const float* ln1_b    = (const float*)d_in[8];
  const float* ff_w1    = (const float*)d_in[9];
  const float* ff_b1    = (const float*)d_in[10];
  const float* ff_w2    = (const float*)d_in[11];
  const float* ff_b2    = (const float*)d_in[12];
  const float* ln2_g    = (const float*)d_in[13];
  const float* ln2_b    = (const float*)d_in[14];
  // d_in[15] = dec_attn_mask (mask computed analytically on-device)

  char* ws = (char*)d_ws;
  const size_t MB = 1u << 20;
  // liveness-overlapped workspace (68 MB total)
  f16*   cat16   = (f16*)(ws + 0);          //  8 MB, dead after QKV GEMM
  f16*   vec16   = (f16*)(ws + 0);          //  4 MB, reuses cat region
  f16*   heads16 = (f16*)(ws + 8 * MB);     // 24 MB, dead after split
  float* attnout = (float*)(ws + 8 * MB);   //  8 MB
  float* outf    = (float*)(ws + 16 * MB);  //  8 MB
  f16*   out16   = (f16*)(ws + 24 * MB);    //  4 MB
  f16*   q16     = (f16*)(ws + 32 * MB);    //  4 MB
  f16*   q16b    = (f16*)(ws + 36 * MB);    //  4 MB
  f16*   k16     = (f16*)(ws + 40 * MB);    //  8 MB
  f16*   vT16    = (f16*)(ws + 48 * MB);    //  8 MB
  f16*   re16    = (f16*)(ws + 56 * MB);    //  4 MB
  f16*   h16     = (f16*)(ws + 32 * MB);    // 16 MB, after attention
  float* core    = (float*)(ws + 48 * MB);  //  8 MB, after attention
  f16*   w16     = (f16*)(ws + 60 * MB);    //  8 MB, sequential weight staging

  const int RB = 2048;                      // Q*B rows
  const int KB = 4096;                      // K*B rows

  // 1) pack cat + convert qkv weights, QKV GEMM -> heads16 (f16)
  build_cat_kernel<<<8192, 256, 0, stream>>>(mems, dec_inp, cat16);
  cvt16_kernel<<<4096, 256, 0, stream>>>(qkv_w, w16, (size_t)3072 * 1024);
  gemm_nt_kernel<<<dim3(3072 / 128, KB / 128), 256, 0, stream>>>(
      cat16, w16, nullptr, nullptr, heads16, KB, 3072, DMODEL, 0);

  // 2) split heads (+r_w_bias), transpose V, pack r_emb
  split_heads_kernel<<<16384, 256, 0, stream>>>(heads16, r_w_bias, q16, q16b, k16, vT16);
  remb_kernel<<<8192, 256, 0, stream>>>(r_emb, re16);

  // 3) fused relative attention (WMMA + 257KB LDS score buffer)
  attn_kernel<<<dim3(QLEN / 64, NH, NB), 128, ATT_SMEM, stream>>>(
      q16, q16b, k16, vT16, re16, r_bias, vec16);

  // 4) output projection
  cvt16_kernel<<<4096, 256, 0, stream>>>(o_w, w16, (size_t)DMODEL * DMODEL);
  gemm_nt_kernel<<<dim3(DMODEL / 128, RB / 128), 256, 0, stream>>>(
      vec16, w16, nullptr, attnout, nullptr, RB, DMODEL, DMODEL, 0);

  // 5) residual + LN1 (also emits f16 for FF)
  add_ln_kernel<<<RB, 256, 0, stream>>>(dec_inp, attnout, ln1_g, ln1_b, outf, out16);

  // 6) FF1 (bias + ReLU, f16 out)
  cvt16_kernel<<<4096, 256, 0, stream>>>(ff_w1, w16, (size_t)DINNER * DMODEL);
  gemm_nt_kernel<<<dim3(DINNER / 128, RB / 128), 256, 0, stream>>>(
      out16, w16, ff_b1, nullptr, h16, RB, DINNER, DMODEL, 1);

  // 7) FF2 (bias, f32 out)
  cvt16_kernel<<<4096, 256, 0, stream>>>(ff_w2, w16, (size_t)DMODEL * DINNER);
  gemm_nt_kernel<<<dim3(DMODEL / 128, RB / 128), 256, 0, stream>>>(
      h16, w16, ff_b2, core, nullptr, RB, DMODEL, DINNER, 0);

  // 8) residual + LN2 -> final output
  add_ln_kernel<<<RB, 256, 0, stream>>>(outf, core, ln2_g, ln2_b, (float*)d_out, nullptr);
}
